// EncoderBlock_81999515616041
// MI455X (gfx1250) — matrix operand order repaired and checked by reading, not compiled, not run on hardware
//
#include <hip/hip_runtime.h>
#include <hip/hip_bf16.h>
#include <stdint.h>

typedef __attribute__((ext_vector_type(16))) _Float16 v16h;
typedef __attribute__((ext_vector_type(8)))  _Float16 v8h;
typedef __attribute__((ext_vector_type(8)))  float    v8f;
typedef __attribute__((ext_vector_type(4)))  unsigned int u32x4;
typedef __attribute__((ext_vector_type(8)))  int          i32x8;
typedef __attribute__((ext_vector_type(4)))  int          i32x4;

#define S_LEN  1024
#define DMODEL 1024

#if defined(__AMDGCN__) && __has_builtin(__builtin_amdgcn_tensor_load_to_lds)
#define HAVE_TDM 1
#else
#define HAVE_TDM 0
#endif

static __device__ __forceinline__ float4 ld4(const float* p) {
    return *reinterpret_cast<const float4*>(p);
}

static __device__ __forceinline__ v16h zero16() {
    v16h z;
    #pragma unroll
    for (int i = 0; i < 16; ++i) z[i] = (_Float16)0.0f;
    return z;
}

static __device__ __forceinline__ v16h cat8(v8h lo, v8h hi) {
    return __builtin_shufflevector(lo, hi, 0,1,2,3,4,5,6,7,8,9,10,11,12,13,14,15);
}

// CDNA5 async copy: 16 bytes global -> LDS, tracked by ASYNCcnt.
static __device__ __forceinline__ void async_b128(unsigned lds_off, const void* gptr) {
    asm volatile("global_load_async_to_lds_b128 %0, %1, off"
                 :: "v"(lds_off), "v"((unsigned long long)(uintptr_t)gptr)
                 : "memory");
}
static __device__ __forceinline__ void wait_async0() {
    asm volatile("s_wait_asynccnt 0x0" ::: "memory");
}
static __device__ __forceinline__ unsigned lds_addr_of(const void* p) {
    // generic pointers to LDS carry the LDS byte offset in the low 32 bits
    return (unsigned)(uintptr_t)p;
}

#if HAVE_TDM
// Tensor Data Mover: 2D f16 tile (tile_w x tile_h elements) from a row-major
// tensor (row stride = stride_elems) into LDS, with pad_amount DWORD-code
// padding inserted every pad_interval-code bytes (reproduces padded LDS rows).
static __device__ __forceinline__ void tdm_load_2d_f16(
    unsigned lds_byte, const void* gptr,
    unsigned tensor_w, unsigned tensor_h, unsigned stride_elems,
    unsigned tile_w, unsigned tile_h,
    unsigned pad_interval_code, unsigned pad_amount_code) {
    const unsigned long long ga = (unsigned long long)(uintptr_t)gptr;
    u32x4 g0;
    g0[0] = 1u;                                            // count=1 (valid D#)
    g0[1] = lds_byte;                                      // lds_addr
    g0[2] = (unsigned)(ga & 0xFFFFFFFFu);                  // global_addr lo
    g0[3] = (unsigned)((ga >> 32) & 0x01FFFFFFu) | (2u << 30);  // addr hi | type=2
    i32x8 g1;
    g1[0] = (int)((1u << 16)                               // data_size=2B
                | (1u << 20)                               // pad_enable
                | (pad_interval_code << 22)
                | (pad_amount_code << 25));
    g1[1] = (int)((tensor_w & 0xFFFFu) << 16);             // tensor_dim0 lo16
    g1[2] = (int)(((tensor_w >> 16) & 0xFFFFu) | ((tensor_h & 0xFFFFu) << 16));
    g1[3] = (int)(((tensor_h >> 16) & 0xFFFFu) | ((tile_w & 0xFFFFu) << 16));
    g1[4] = (int)(tile_h & 0xFFFFu);                       // tile_dim1; tile_dim2=0
    g1[5] = (int)stride_elems;                             // tensor_dim0_stride lo
    g1[6] = 0;
    g1[7] = 0;
    const i32x4 gz = {0, 0, 0, 0};
#if __clang_major__ >= 23
    const i32x8 gz8 = {0, 0, 0, 0, 0, 0, 0, 0};
    __builtin_amdgcn_tensor_load_to_lds(g0, g1, gz, gz, gz8, 0);
#else
    __builtin_amdgcn_tensor_load_to_lds(g0, g1, gz, gz, 0);
#endif
}
#endif

// ---------------------------------------------------------------------------
// f32 -> f16 elementwise conversion (one-time, amortized against 23.3 TB/s)
// ---------------------------------------------------------------------------
__global__ __launch_bounds__(256)
void cvt_kernel(const float* __restrict__ in, _Float16* __restrict__ out, int n) {
    int i = (blockIdx.x * 256 + threadIdx.x) * 8;
    if (i >= n) return;
    float4 a = ld4(in + i), b = ld4(in + i + 4);
    v8h o;
    o[0]=(_Float16)a.x; o[1]=(_Float16)a.y; o[2]=(_Float16)a.z; o[3]=(_Float16)a.w;
    o[4]=(_Float16)b.x; o[5]=(_Float16)b.y; o[6]=(_Float16)b.z; o[7]=(_Float16)b.w;
    *reinterpret_cast<v8h*>(out + i) = o;
}

// ---------------------------------------------------------------------------
// TDM/async-LDS staged WMMA GEMM: C[M,N] = A[M,K]@W[K,N] + bias (+resid)(ReLU)
// A,W f16; output f32 (Cf) or f16 (Ch). 256 thr = 8 waves (4M x 2N),
// WG tile 128x128, wave tile 32x64 -> 8 wmma per K-step of 32.
// Double-buffered LDS tiles filled by the Tensor Data Mover (wave 0 issues a
// 2D descriptor per tile; padded rows via D# pad fields), fallback per-lane
// global_load_async_to_lds_b128.
// ---------------------------------------------------------------------------
__global__ __launch_bounds__(256)
void gemm_async_wmma(const _Float16* __restrict__ A, const _Float16* __restrict__ W,
                     const float* __restrict__ bias, const float* __restrict__ resid,
                     float* __restrict__ Cf, _Float16* __restrict__ Ch,
                     int M, int N, int Kd, int relu) {
    __shared__ _Float16 As[2][128][40];    // 128 rows x 32 K (+8 pad = 16B)
    __shared__ _Float16 Bs[2][32][136];    // 32 K rows x 128 N (+8 pad = 16B)

    const int lane = threadIdx.x & 31;
    const int wave = threadIdx.x >> 5;
    const int half = lane >> 4;
    const int lm   = lane & 15;
    const int wm   = wave >> 1;            // 0..3 (M)
    const int wn   = wave & 1;             // 0..1 (N)
    const int m0   = blockIdx.y * 128;
    const int n0   = blockIdx.x * 128;
    const int mw   = wm * 32;
    const int nw   = wn * 64;

    v8f acc[2][4];
    #pragma unroll
    for (int i = 0; i < 2; ++i)
        #pragma unroll
        for (int j = 0; j < 4; ++j) acc[i][j] = {};

    auto fill = [&](int buf, int k0) {
#if HAVE_TDM
        if (wave == 0) {
            // A tile: rows of 32 f16 = 64B -> pad 16B every 64B (codes 3, 3)
            tdm_load_2d_f16(lds_addr_of(&As[buf][0][0]),
                            A + (size_t)m0 * Kd + k0,
                            (unsigned)Kd, (unsigned)M, (unsigned)Kd,
                            32u, 128u, 3u, 3u);
            // B tile: rows of 128 f16 = 256B -> pad 16B every 256B (codes 5, 3)
            tdm_load_2d_f16(lds_addr_of(&Bs[buf][0][0]),
                            W + (size_t)k0 * N + n0,
                            (unsigned)N, (unsigned)Kd, (unsigned)N,
                            128u, 32u, 5u, 3u);
        }
#else
        #pragma unroll
        for (int p = 0; p < 2; ++p) {                 // A: 512 x 16B chunks
            const int idx = threadIdx.x + p * 256;
            const int row = idx >> 2;
            const int c   = idx & 3;
            async_b128(lds_addr_of(&As[buf][row][c * 8]),
                       A + (size_t)(m0 + row) * Kd + k0 + c * 8);
        }
        #pragma unroll
        for (int p = 0; p < 2; ++p) {                 // B: 512 x 16B chunks
            const int idx = threadIdx.x + p * 256;
            const int row = idx >> 4;
            const int c   = idx & 15;
            async_b128(lds_addr_of(&Bs[buf][row][c * 8]),
                       W + (size_t)(k0 + row) * N + n0 + c * 8);
        }
#endif
    };

    const int nk = Kd >> 5;
    fill(0, 0);
    for (int kb = 0; kb < nk; ++kb) {
#if HAVE_TDM
        if (wave == 0) __builtin_amdgcn_s_wait_tensorcnt(0);
#else
        wait_async0();
#endif
        __syncthreads();
        if (kb + 1 < nk) fill((kb + 1) & 1, (kb + 1) << 5);
        const int buf = kb & 1;

        v16h afr[2];
        #pragma unroll
        for (int i = 0; i < 2; ++i) {
            const _Float16* ar = &As[buf][mw + i * 16 + lm][0];
            afr[i] = cat8(*reinterpret_cast<const v8h*>(ar + 8 * half),
                          *reinterpret_cast<const v8h*>(ar + 16 + 8 * half));
        }
        v16h bfr[4];
        #pragma unroll
        for (int j = 0; j < 4; ++j) {
            #pragma unroll
            for (int e = 0; e < 16; ++e)
                bfr[j][e] = Bs[buf][(e < 8 ? 8 * half + e : 16 + 8 * half + (e - 8))][nw + j * 16 + lm];
        }
        #pragma unroll
        for (int i = 0; i < 2; ++i)
            #pragma unroll
            for (int j = 0; j < 4; ++j)
                acc[i][j] = __builtin_amdgcn_wmma_f32_16x16x32_f16(
                    false, afr[i], false, bfr[j], (short)0, acc[i][j], false, false);
    }

    #pragma unroll
    for (int i = 0; i < 2; ++i) {
        #pragma unroll
        for (int j = 0; j < 4; ++j) {
            const int ng = n0 + nw + j * 16 + lm;
            const float bsv = bias[ng];
            #pragma unroll
            for (int r = 0; r < 8; ++r) {
                const int mg = m0 + mw + i * 16 + r + 8 * half;
                float v = acc[i][j][r] + bsv;
                if (resid) v += resid[(size_t)mg * N + ng];
                if (relu)  v = fmaxf(v, 0.0f);
                if (Cf) Cf[(size_t)mg * N + ng] = v;
                if (Ch) Ch[(size_t)mg * N + ng] = (_Float16)v;
            }
        }
    }
}

// ---------------------------------------------------------------------------
// Flash attention, 64 "heads" of dim 16, scale=4 folded into Q (f16 inputs).
// grid = (S/128, 64, B); 8 waves, each owns 16 query rows. Per 16-key block:
// K tile staged transposed in LDS, V tile staged via async-to-LDS, S-tile and
// P@V via WMMA (K padded 16->32), online softmax via 16-lane shfl reductions.
// ---------------------------------------------------------------------------
__global__ __launch_bounds__(256)
void attn_wmma_kernel(const _Float16* __restrict__ Q, const _Float16* __restrict__ Kh,
                      const _Float16* __restrict__ Vh, const unsigned char* __restrict__ msk,
                      float* __restrict__ O) {
    __shared__ _Float16 pbuf[8][16][16];   // per-wave P bounce buffer
    __shared__ _Float16 Kst[16][16];       // K tile, stored transposed [d][m]
    __shared__ _Float16 Vs[16][16];        // V tile, natural [m][d]

    const int lane = threadIdx.x & 31;
    const int wave = threadIdx.x >> 5;
    const int half = lane >> 4;
    const int lm   = lane & 15;
    const int t    = threadIdx.x;
    const int b    = blockIdx.z;
    const int h    = blockIdx.y;
    const int n0   = blockIdx.x * 128 + wave * 16;

    // Q fragment (A layout), scaled by 4 (exact in f16); K elems 16..31 zero.
    const _Float16* qrow = Q + ((size_t)b * S_LEN + n0 + lm) * DMODEL + h * 16 + 8 * half;
    v8h qv = *reinterpret_cast<const v8h*>(qrow);
    v16h aq = zero16();
    #pragma unroll
    for (int e = 0; e < 8; ++e) aq[e] = qv[e] * (_Float16)4.0f;

    v8f o = {};
    float mrun[8], lrun[8];
    #pragma unroll
    for (int r = 0; r < 8; ++r) { mrun[r] = -__builtin_inff(); lrun[r] = 0.0f; }

    const int srow = t >> 4, scol = t & 15;   // staging coords (256 threads)

    for (int m0 = 0; m0 < S_LEN; m0 += 16) {
        // stage K (transposed) and V tiles for this (b,h,m-block)
        Kst[scol][srow] = Kh[((size_t)b * S_LEN + m0 + srow) * DMODEL + h * 16 + scol];
        if (t < 32) {   // V: 16 rows x 32B = 32 async b128 chunks (wave 0)
            const int vr = t >> 1, vc = t & 1;
            async_b128(lds_addr_of(&Vs[vr][vc * 8]),
                       Vh + ((size_t)b * S_LEN + m0 + vr) * DMODEL + h * 16 + vc * 8);
        }
        wait_async0();
        __syncthreads();

        // K fragment (B layout): lane = m col, element e<8 = d row 8*half+e
        v16h bk = zero16();
        #pragma unroll
        for (int e = 0; e < 8; ++e) bk[e] = Kst[8 * half + e][lm];
        v8f s = {};
        s = __builtin_amdgcn_wmma_f32_16x16x32_f16(false, aq, false, bk,
                                                   (short)0, s, false, false);

        // mask: -inf where mask[b, n, m]
        const unsigned char* mbase = msk + ((size_t)b * S_LEN + n0) * S_LEN + m0 + lm;
        #pragma unroll
        for (int r = 0; r < 8; ++r) {
            if (mbase[(size_t)(r + 8 * half) * S_LEN]) s[r] = -__builtin_inff();
        }

        // online softmax (rows across the 16 lanes of each half)
        #pragma unroll
        for (int r = 0; r < 8; ++r) {
            float v = s[r];
            #pragma unroll
            for (int off = 1; off < 16; off <<= 1)
                v = fmaxf(v, __shfl_xor(v, off, 16));
            const float mnew  = fmaxf(mrun[r], v);
            const float alpha = __expf(mrun[r] - mnew);
            const float p     = __expf(s[r] - mnew);
            float rs = p;
            #pragma unroll
            for (int off = 1; off < 16; off <<= 1)
                rs += __shfl_xor(rs, off, 16);
            lrun[r] = lrun[r] * alpha + rs;
            mrun[r] = mnew;
            o[r]    = o[r] * alpha;
            pbuf[wave][r + 8 * half][lm] = (_Float16)p;
        }

        // P as A fragment (per-wave LDS ops stay in order within the wave)
        v8h pv = *reinterpret_cast<const v8h*>(&pbuf[wave][lm][8 * half]);
        v16h ap = zero16();
        #pragma unroll
        for (int e = 0; e < 8; ++e) ap[e] = pv[e];

        // V fragment (B layout): lane = d col, element e<8 = m row 8*half+e
        v16h bvf = zero16();
        #pragma unroll
        for (int e = 0; e < 8; ++e) bvf[e] = Vs[8 * half + e][lm];
        o = __builtin_amdgcn_wmma_f32_16x16x32_f16(false, ap, false, bvf,
                                                   (short)0, o, false, false);
        __syncthreads();   // protect Kst/Vs before next stage
    }

    #pragma unroll
    for (int r = 0; r < 8; ++r) {
        const int row = r + 8 * half;
        O[((size_t)b * S_LEN + n0 + row) * DMODEL + h * 16 + lm] = o[r] / lrun[r];
    }
}

// ---------------------------------------------------------------------------
// LayerNorm over rows of 1024, fused residual add; optional extra f16 output.
// ---------------------------------------------------------------------------
__global__ __launch_bounds__(256)
void ln_kernel(const float* __restrict__ x, const float* __restrict__ res,
               const float* __restrict__ g, const float* __restrict__ bb,
               float* __restrict__ out, _Float16* __restrict__ outh) {
    __shared__ float ssum[8], ssq[8];
    const int row  = blockIdx.x;
    const int t    = threadIdx.x;
    const int lane = t & 31;
    const int wave = t >> 5;

    float4 v = ld4(x + (size_t)row * DMODEL + t * 4);
    if (res) {
        float4 u = ld4(res + (size_t)row * DMODEL + t * 4);
        v.x += u.x; v.y += u.y; v.z += u.z; v.w += u.w;
    }
    float s = v.x + v.y + v.z + v.w;
    float q = v.x * v.x + v.y * v.y + v.z * v.z + v.w * v.w;
    #pragma unroll
    for (int off = 1; off < 32; off <<= 1) {
        s += __shfl_xor(s, off, 32);
        q += __shfl_xor(q, off, 32);
    }
    if (lane == 0) { ssum[wave] = s; ssq[wave] = q; }
    __syncthreads();
    float St = 0.f, Qt = 0.f;
    #pragma unroll
    for (int i = 0; i < 8; ++i) { St += ssum[i]; Qt += ssq[i]; }
    const float mean = St * (1.0f / DMODEL);
    const float var  = Qt * (1.0f / DMODEL) - mean * mean;
    const float rstd = rsqrtf(var + 1e-5f);

    float4 gg = ld4(g + t * 4);
    float4 be = ld4(bb + t * 4);
    float4 ov;
    ov.x = (v.x - mean) * rstd * gg.x + be.x;
    ov.y = (v.y - mean) * rstd * gg.y + be.y;
    ov.z = (v.z - mean) * rstd * gg.z + be.z;
    ov.w = (v.w - mean) * rstd * gg.w + be.w;
    *reinterpret_cast<float4*>(out + (size_t)row * DMODEL + t * 4) = ov;
    if (outh) {
        _Float16* oh = outh + (size_t)row * DMODEL + t * 4;
        oh[0] = (_Float16)ov.x; oh[1] = (_Float16)ov.y;
        oh[2] = (_Float16)ov.z; oh[3] = (_Float16)ov.w;
    }
}

// ---------------------------------------------------------------------------
extern "C" void kernel_launch(void* const* d_in, const int* in_sizes, int n_in,
                              void* d_out, int out_size, void* d_ws, size_t ws_size,
                              hipStream_t stream) {
    (void)in_sizes; (void)n_in; (void)out_size; (void)ws_size;

    const float* x   = (const float*)d_in[0];
    const unsigned char* msk = (const unsigned char*)d_in[1];
    const float* Wq = (const float*)d_in[2];
    const float* bq = (const float*)d_in[3];
    const float* Wk = (const float*)d_in[4];
    const float* bk = (const float*)d_in[5];
    const float* Wv = (const float*)d_in[6];
    const float* bv = (const float*)d_in[7];
    const float* g1 = (const float*)d_in[8];
    const float* e1 = (const float*)d_in[9];
    const float* W1 = (const float*)d_in[10];
    const float* b1 = (const float*)d_in[11];
    const float* W2 = (const float*)d_in[12];
    const float* b2 = (const float*)d_in[13];
    const float* g2 = (const float*)d_in[14];
    const float* e2 = (const float*)d_in[15];
    float* out = (float*)d_out;

    const size_t TOK = (size_t)2 * S_LEN * DMODEL;   // 2M elements
    const size_t WSM = (size_t)DMODEL * DMODEL;      // 1M
    const size_t WFF = (size_t)DMODEL * 4096;        // 4M
    char* w = (char*)d_ws;
    auto carve = [&](size_t bytes) { char* p = w; w += (bytes + 255) & ~(size_t)255; return p; };

    float*    attn = (float*)carve(TOK * 4);
    float*    h1   = (float*)carve(TOK * 4);
    float*    ff2  = (float*)carve(TOK * 4);
    _Float16* Xh   = (_Float16*)carve(TOK * 2);
    _Float16* Wqh  = (_Float16*)carve(WSM * 2);
    _Float16* Wkh  = (_Float16*)carve(WSM * 2);
    _Float16* Wvh  = (_Float16*)carve(WSM * 2);
    _Float16* W1h  = (_Float16*)carve(WFF * 2);
    _Float16* W2h  = (_Float16*)carve(WFF * 2);
    _Float16* Qh   = (_Float16*)carve(TOK * 2);
    _Float16* Kb   = (_Float16*)carve(TOK * 2);
    _Float16* Vb   = (_Float16*)carve(TOK * 2);
    _Float16* h1h  = (_Float16*)carve(TOK * 2);
    _Float16* ff1h = (_Float16*)carve((size_t)2048 * 4096 * 2);

    const dim3 blk(256);
    const int M = 2 * S_LEN;   // 2048 token rows

    // one-time f32 -> f16 conversions
    cvt_kernel<<<dim3((int)(TOK / 2048)), blk, 0, stream>>>(x,  Xh,  (int)TOK);
    cvt_kernel<<<dim3((int)(WSM / 2048)), blk, 0, stream>>>(Wq, Wqh, (int)WSM);
    cvt_kernel<<<dim3((int)(WSM / 2048)), blk, 0, stream>>>(Wk, Wkh, (int)WSM);
    cvt_kernel<<<dim3((int)(WSM / 2048)), blk, 0, stream>>>(Wv, Wvh, (int)WSM);
    cvt_kernel<<<dim3((int)(WFF / 2048)), blk, 0, stream>>>(W1, W1h, (int)WFF);
    cvt_kernel<<<dim3((int)(WFF / 2048)), blk, 0, stream>>>(W2, W2h, (int)WFF);

    // Q/K/V projections (f16 outputs feed WMMA attention)
    gemm_async_wmma<<<dim3(DMODEL / 128, M / 128), blk, 0, stream>>>(
        Xh, Wqh, bq, nullptr, nullptr, Qh, M, DMODEL, DMODEL, 0);
    gemm_async_wmma<<<dim3(DMODEL / 128, M / 128), blk, 0, stream>>>(
        Xh, Wkh, bk, nullptr, nullptr, Kb, M, DMODEL, DMODEL, 0);
    gemm_async_wmma<<<dim3(DMODEL / 128, M / 128), blk, 0, stream>>>(
        Xh, Wvh, bv, nullptr, nullptr, Vb, M, DMODEL, DMODEL, 0);

    // attention
    attn_wmma_kernel<<<dim3(S_LEN / 128, 64, 2), blk, 0, stream>>>(
        Qh, Kb, Vb, msk, attn);

    // h1 = LN(attn + x)   (f32 for residual, f16 for next GEMM)
    ln_kernel<<<dim3(M), blk, 0, stream>>>(attn, x, g1, e1, h1, h1h);

    // ff1 = relu(h1 @ W1 + b1)  (f16 out)
    gemm_async_wmma<<<dim3(4096 / 128, M / 128), blk, 0, stream>>>(
        h1h, W1h, b1, nullptr, nullptr, ff1h, M, 4096, DMODEL, 1);

    // ff2 = ff1 @ W2 + b2 + h1  (f32 out)
    gemm_async_wmma<<<dim3(DMODEL / 128, M / 128), blk, 0, stream>>>(
        ff1h, W2h, b2, h1, ff2, nullptr, M, DMODEL, 4096, 0);

    // out = LN(ff2)
    ln_kernel<<<dim3(M), blk, 0, stream>>>(ff2, nullptr, g2, e2, out, nullptr);
}
